// LAME_20650202759384
// MI455X (gfx1250) — compile-verified, matrix operand order checked by reading, not run on hardware
//
#include <hip/hip_runtime.h>
#include <hip/hip_bf16.h>
#include <cstddef>
#include <cstdint>

// ---------------------------------------------------------------------------
// LAME on gfx1250: Gram-matrix KNN via v_wmma_f32_16x16x32_f16 with B tiles
// streamed by the Tensor Data Mover (TENSOR_LOAD_TO_LDS, double-buffered),
// then 100 gather/softmax fixed-point iterations (all L2-resident).
// ---------------------------------------------------------------------------

typedef __attribute__((ext_vector_type(16))) _Float16     v16h;
typedef __attribute__((ext_vector_type(8)))  float        v8f;
typedef __attribute__((ext_vector_type(4)))  unsigned int v4u;
typedef __attribute__((ext_vector_type(8)))  int          v8i;
typedef __attribute__((ext_vector_type(4)))  int          v4i;

#define N_ROWS   8192
#define N_FEAT   256
#define N_CLS    64
#define NWAVES   4
#define ROWS_WG  (NWAVES * 16)      // 64 rows per workgroup
#define NCOLT    (N_ROWS / 16)      // 512 column tiles
#define BSTRIDE  264                // f16 elems per padded B-tile row (256+8)
#define BTILE_B  (16 * BSTRIDE * 2) // 8448 bytes per B tile in LDS

// -------------------------- kernel 1: normalize ----------------------------
__global__ void lame_normalize(const float* __restrict__ feats,
                               _Float16* __restrict__ fn) {
  __shared__ float red[64];
  const int row = blockIdx.x;
  const int t   = threadIdx.x;                      // 0..63, 4 elems each
  const float4 v = reinterpret_cast<const float4*>(feats + (size_t)row * N_FEAT)[t];
  float ss = v.x * v.x + v.y * v.y + v.z * v.z + v.w * v.w;
  red[t] = ss; __syncthreads();
  for (int s = 32; s > 0; s >>= 1) { if (t < s) red[t] += red[t + s]; __syncthreads(); }
  const float nrm   = sqrtf(red[0]);
  const float scale = 1.0f / fmaxf(nrm, 1e-12f);
  const size_t base = (size_t)row * N_FEAT + 4 * t;
  fn[base + 0] = (_Float16)(v.x * scale);
  fn[base + 1] = (_Float16)(v.y * scale);
  fn[base + 2] = (_Float16)(v.z * scale);
  fn[base + 3] = (_Float16)(v.w * scale);
}

// ----------------------- kernel 2: unary + Y0 = softmax --------------------
__global__ void lame_unary_y0(const float* __restrict__ scores,
                              float* __restrict__ unary,
                              float* __restrict__ Y0) {
  __shared__ float red[64];
  const int row = blockIdx.x;
  const int c   = threadIdx.x;
  const float s = scores[(size_t)row * N_CLS + c];
  const float u = -logf(s + 1e-10f);
  unary[(size_t)row * N_CLS + c] = u;
  const float z = -u;
  red[c] = z; __syncthreads();
  for (int st = 32; st > 0; st >>= 1) { if (c < st) red[c] = fmaxf(red[c], red[c + st]); __syncthreads(); }
  const float m = red[0]; __syncthreads();
  const float e = expf(z - m);
  red[c] = e; __syncthreads();
  for (int st = 32; st > 0; st >>= 1) { if (c < st) red[c] += red[c + st]; __syncthreads(); }
  Y0[(size_t)row * N_CLS + c] = e / red[0];
}

// ------------------- kernel 3: WMMA Gram + fused top-6 KNN -----------------
__device__ __forceinline__ bool knn_better(float v, int i, float tv, int ti) {
  return (v > tv) || (v == tv && i < ti);            // jax.lax.top_k tie rule
}

// Issue one TDM 2D-tile load: 16 rows x 256 f16 from fn + tile*16 rows into
// LDS at lds_byte_addr, with 16B padding appended per 512B row (-> BSTRIDE).
__device__ __forceinline__ void tdm_load_btile(const _Float16* fn,
                                               unsigned lds_byte_addr,
                                               int tile) {
  const unsigned long long gaddr =
      (unsigned long long)(uintptr_t)fn +
      (unsigned long long)tile * (16ull * N_FEAT * 2ull);
  v4u g0;
  g0[0] = 1u;                                        // count=1, user descriptor
  g0[1] = lds_byte_addr;                             // lds_addr
  g0[2] = (unsigned)(gaddr & 0xffffffffull);         // global_addr[31:0]
  g0[3] = (unsigned)((gaddr >> 32) & 0x1ffffffull)   // global_addr[56:32]
          | 0x80000000u;                             // type = 2 ("image")
  v8i g1;
  g1[0] = (1 << 16)                                  // data_size = 2 bytes
        | (1 << 20)                                  // pad_enable
        | (6 << 22)                                  // pad_interval: 128 DW (512B)
        | (3 << 25);                                 // pad_amount: 4 DW (16B)
  g1[1] = (N_FEAT & 0xffff) << 16;                   // tensor_dim0 lo16 = 256
  g1[2] = (N_ROWS & 0xffff) << 16;                   // tensor_dim1 lo16 = 8192
  g1[3] = (N_FEAT & 0xffff) << 16;                   // tile_dim0 = 256
  g1[4] = 16;                                        // tile_dim1 = 16, tile_dim2 = 0
  g1[5] = N_FEAT;                                    // tensor_dim0_stride = 256
  g1[6] = 0;
  g1[7] = 0;
  const v4i zg4 = {0, 0, 0, 0};                      // 2D tensor: groups 2/3 unused
  const v8i zg8 = {0, 0, 0, 0, 0, 0, 0, 0};          // trailing group unused
  __builtin_amdgcn_tensor_load_to_lds(g0, g1, zg4, zg4, zg8, 0);
}

__global__ void __launch_bounds__(NWAVES * 32)
lame_knn(const _Float16* __restrict__ fn, int* __restrict__ nbr) {
  __shared__ _Float16 Bsh[2][16 * BSTRIDE];          // 2 x 8448 B (TDM dest)
  __shared__ float    tileF[NWAVES][16][17];         // C-tile transpose
  __shared__ float    mV[NWAVES][16][2][6];
  __shared__ int      mI[NWAVES][16][2][6];

  const int tid  = threadIdx.x;
  const int wave = tid >> 5;
  const int lane = tid & 31;
  const int sub  = lane & 15;                        // M (A) / N (B) index
  const int grp  = lane >> 4;                        // K half selector
  const int myrow = blockIdx.x * ROWS_WG + wave * 16 + sub;
  const unsigned lds_b0 = (unsigned)(uintptr_t)(&Bsh[0][0]);

  // A fragments resident in VGPRs: K = 256 -> 8 fragments of K=32.
  // ISA layout: lanes 0-15 hold K {k0..k0+7, k0+16..k0+23}, lanes 16-31 hold
  // K {k0+8..k0+15, k0+24..k0+31}. Gram symmetry: same loader serves A and B.
  v16h a[8];
#pragma unroll
  for (int kk = 0; kk < 8; ++kk) {
    const int kb = kk * 32 + grp * 8;
    union { v16h v; float4 q[2]; } u;
    u.q[0] = *reinterpret_cast<const float4*>(fn + (size_t)myrow * N_FEAT + kb);
    u.q[1] = *reinterpret_cast<const float4*>(fn + (size_t)myrow * N_FEAT + kb + 16);
    a[kk] = u.v;
  }

  float tv[6]; int ti[6];
#pragma unroll
  for (int k = 0; k < 6; ++k) { tv[k] = -3.0e38f; ti[k] = 0x7fffffff; }

  // Scalar (wave-uniform) guard so only wave 0 drives the TDM: TDM ignores
  // EXEC, so the branch itself must be scalar, not a lane mask.
  const bool tdm_wave = (__builtin_amdgcn_readfirstlane(wave) == 0);

  if (tdm_wave) tdm_load_btile(fn, lds_b0, 0);       // prologue: tile 0 -> buf 0

  for (int t = 0; t < NCOLT; ++t) {
    if (tdm_wave) {
      if (t + 1 < NCOLT) {
        tdm_load_btile(fn, lds_b0 + (unsigned)(((t + 1) & 1) * BTILE_B), t + 1);
        __builtin_amdgcn_s_wait_tensorcnt(1);        // tile t complete
      } else {
        __builtin_amdgcn_s_wait_tensorcnt(0);
      }
    }
    __syncthreads();                                 // buf[t&1] visible to all

    const _Float16* Bp = &Bsh[t & 1][0];
    v8f c = {};
#pragma unroll
    for (int kk = 0; kk < 8; ++kk) {
      const int kb = kk * 32 + grp * 8;
      union { v16h v; float4 q[2]; } u;
      u.q[0] = *reinterpret_cast<const float4*>(Bp + sub * BSTRIDE + kb);
      u.q[1] = *reinterpret_cast<const float4*>(Bp + sub * BSTRIDE + kb + 16);
      c = __builtin_amdgcn_wmma_f32_16x16x32_f16(false, a[kk], false, u.v,
                                                 (short)0, c, false, false);
    }

    // C layout: VGPR r = element (M = r + 8*grp, N = sub). Transpose via LDS;
    // same-wave LDS ops are hardware in-order, wave_barrier pins compiler order.
#pragma unroll
    for (int r = 0; r < 8; ++r) tileF[wave][r + 8 * grp][sub] = c[r];
    __builtin_amdgcn_wave_barrier();

    // Fused selection: lane (sub,grp) scans row `sub`, cols grp*8 .. grp*8+7.
    const int C0 = t << 4;
#pragma unroll
    for (int j = 0; j < 8; ++j) {
      const int   n  = grp * 8 + j;
      const float v  = tileF[wave][sub][n];
      const int   id = C0 + n;
      if (knn_better(v, id, tv[5], ti[5])) {
        tv[5] = v; ti[5] = id;
#pragma unroll
        for (int k = 5; k > 0; --k) {
          if (knn_better(tv[k], ti[k], tv[k - 1], ti[k - 1])) {
            const float fv = tv[k]; tv[k] = tv[k - 1]; tv[k - 1] = fv;
            const int   ii = ti[k]; ti[k] = ti[k - 1]; ti[k - 1] = ii;
          }
        }
      }
    }
    __syncthreads();                                 // all done reading buf[t&1]
  }

  // Merge the two lane-group partial lists per row, drop rank 0 (self).
#pragma unroll
  for (int k = 0; k < 6; ++k) { mV[wave][sub][grp][k] = tv[k]; mI[wave][sub][grp][k] = ti[k]; }
  __syncthreads();
  if (grp == 0) {
    float fv[6]; int fi[6];
#pragma unroll
    for (int k = 0; k < 6; ++k) { fv[k] = mV[wave][sub][0][k]; fi[k] = mI[wave][sub][0][k]; }
#pragma unroll
    for (int k = 0; k < 6; ++k) {
      const float v  = mV[wave][sub][1][k];
      const int   id = mI[wave][sub][1][k];
      if (knn_better(v, id, fv[5], fi[5])) {
        fv[5] = v; fi[5] = id;
#pragma unroll
        for (int q = 5; q > 0; --q) {
          if (knn_better(fv[q], fi[q], fv[q - 1], fi[q - 1])) {
            const float tf = fv[q]; fv[q] = fv[q - 1]; fv[q - 1] = tf;
            const int   tii = fi[q]; fi[q] = fi[q - 1]; fi[q - 1] = tii;
          }
        }
      }
    }
#pragma unroll
    for (int k = 0; k < 5; ++k) nbr[(size_t)myrow * 5 + k] = fi[k + 1];
  }
}

// ------------- kernel 4: one fixed-point iteration (gather+softmax) --------
__global__ void lame_iter(const float* __restrict__ unary,
                          const int* __restrict__ nbr,
                          const float* __restrict__ Ysrc,
                          float* __restrict__ Ydst) {
  __shared__ float red[64];
  const int row = blockIdx.x;
  const int c   = threadIdx.x;
  float pw = 0.0f;
#pragma unroll
  for (int j = 0; j < 5; ++j) {
    const int nb = nbr[(size_t)row * 5 + j];
    pw += Ysrc[(size_t)nb * N_CLS + c];
  }
  const float z = pw - unary[(size_t)row * N_CLS + c];   // LAM = 1
  red[c] = z; __syncthreads();
  for (int st = 32; st > 0; st >>= 1) { if (c < st) red[c] = fmaxf(red[c], red[c + st]); __syncthreads(); }
  const float m = red[0]; __syncthreads();
  const float e = expf(z - m);
  red[c] = e; __syncthreads();
  for (int st = 32; st > 0; st >>= 1) { if (c < st) red[c] += red[c + st]; __syncthreads(); }
  Ydst[(size_t)row * N_CLS + c] = e / red[0];
}

// ---------------------------------------------------------------------------
extern "C" void kernel_launch(void* const* d_in, const int* in_sizes, int n_in,
                              void* d_out, int out_size, void* d_ws, size_t ws_size,
                              hipStream_t stream) {
  const float* scores = (const float*)d_in[0];   // (8192, 64)
  const float* feats  = (const float*)d_in[1];   // (8192, 256)
  float* out = (float*)d_out;                    // (8192, 64)

  char* ws = (char*)d_ws;
  _Float16* fn    = (_Float16*)(ws);                         // 4 MiB
  float*    unary = (float*)(ws + 4194304);                  // 2 MiB
  float*    Ya    = (float*)(ws + 6291456);                  // 2 MiB
  float*    Yb    = (float*)(ws + 8388608);                  // 2 MiB
  int*      nbr   = (int*)(ws + 10485760);                   // 160 KiB

  lame_normalize<<<N_ROWS, 64, 0, stream>>>(feats, fn);
  lame_unary_y0 <<<N_ROWS, 64, 0, stream>>>(scores, unary, Ya);
  lame_knn      <<<N_ROWS / ROWS_WG, NWAVES * 32, 0, stream>>>(fn, nbr);

  float* src = Ya; float* dst = Yb;
  for (int it = 0; it < 100; ++it) {
    float* d = (it == 99) ? out : dst;
    lame_iter<<<N_ROWS, 64, 0, stream>>>(unary, nbr, src, d);
    float* tmp = src; src = dst; dst = tmp;
  }
}
